// Origin_Interaction_8297876816074
// MI455X (gfx1250) — matrix operand order repaired and checked by reading, compiled, not run on hardware
//
#include <hip/hip_runtime.h>

// CDNA5 / gfx1250 attention kernel set for Origin_Interaction.
// f16 WMMA (v_wmma_f32_16x16x32_f16) for QK^T and PV; fp32 softmax;
// local CLIP x CLIP term folded as a 3x3 weighted sum over raw scores.

typedef __attribute__((ext_vector_type(16))) _Float16 v16h;
typedef __attribute__((ext_vector_type(8)))  _Float16 v8h;
typedef __attribute__((ext_vector_type(8)))  float    v8f;
typedef __attribute__((ext_vector_type(4)))  float    v4f;

#define B_    16
#define L_    768
#define D_    128
#define NBLK  256          // L_/3 block columns
#define ROWS_PER_WG 48     // lcm(16, 3): 3 WMMA row tiles, 16 CLIP blocks
#define INV_T 0.08838834764831845f  // 1/sqrt(128)

// ---- WMMA fragment loaders (wave32, 16x16x32 f16) ----
// A (16x32, row major, ld halves): lane<16 holds row m=lane, K = 0..15;
// lane>=16 holds row m=lane-16, K = 16..31.
__device__ __forceinline__ v16h load_a_f16(const _Float16* base, int ld, int row,
                                           int k0, int lane) {
    int m = lane & 15, g = lane >> 4;
    const _Float16* p = base + (size_t)(row + m) * ld + k0 + g * 16;
    v8h lo = *(const v8h*)p;          // K = g*16 .. g*16+7
    v8h hi = *(const v8h*)(p + 8);    // K = g*16+8 .. g*16+15
    v16h a;
#pragma unroll
    for (int i = 0; i < 8; ++i) { a[i] = lo[i]; a[i + 8] = hi[i]; }
    return a;
}

// B (32x16): source stored as 16 rows of K-contiguous halves (K^T / V^T form).
// lanes 0-15 hold column n = lane, K = 0..15; lanes 16-31 hold K = 16..31.
__device__ __forceinline__ v16h load_b_f16(const _Float16* base, int ld, int col,
                                           int k0, int lane) {
    int n = lane & 15, g = lane >> 4;
    return *(const v16h*)(base + (size_t)(col + n) * ld + k0 + g * 16);
}

// ---- Kernel 0: precision/layout prep: q,k -> f16 row major; v -> f16 transposed ----
__global__ void __launch_bounds__(256) prep_kernel(
    const float* __restrict__ q, const float* __restrict__ k,
    const float* __restrict__ v,
    _Float16* __restrict__ qh, _Float16* __restrict__ kh,
    _Float16* __restrict__ vt)
{
    int idx = blockIdx.x * 256 + threadIdx.x;   // over B*L*D
    qh[idx] = (_Float16)q[idx];
    kh[idx] = (_Float16)k[idx];
    int d = idx & (D_ - 1);
    int l = (idx >> 7) % L_;
    int b = idx / (L_ * D_);
    vt[(size_t)b * D_ * L_ + (size_t)d * L_ + l] = (_Float16)v[idx];
}

// ---- Kernel 1: S = QK^T panel (48x768) in LDS, local 3x3 fold, softmax, attn out ----
__global__ void __launch_bounds__(256) attn_scores_softmax_kernel(
    const _Float16* __restrict__ qh, const _Float16* __restrict__ kh,
    const float* __restrict__ fc_w, const float* __restrict__ fc_b,
    float* __restrict__ attn_out)
{
    extern __shared__ float smem[];
    float* Sbuf = smem;                       // 48 * 768 f32 = 144 KB
    float* Lbuf = smem + ROWS_PER_WG * L_;    // 16 * 256 f32 =  16 KB

    const int wg = blockIdx.x;
    const int b  = wg >> 4;
    const int rowbase = (wg & 15) * ROWS_PER_WG;

    const int tid  = threadIdx.x;
    const int lane = tid & 31;
    const int wid  = tid >> 5;

    const _Float16* qb = qh + (size_t)b * L_ * D_;
    const _Float16* kb = kh + (size_t)b * L_ * D_;

    // Phase 1: raw scores. 8 waves x (3 row tiles x 6 col tiles) of 16x16.
    for (int rt = 0; rt < 3; ++rt) {
        v16h afrag[4];
#pragma unroll
        for (int kk = 0; kk < 4; ++kk)
            afrag[kk] = load_a_f16(qb, D_, rowbase + rt * 16, kk * 32, lane);
        for (int c6 = 0; c6 < 6; ++c6) {
            int ct = wid * 6 + c6;
            v8f acc = {};
#pragma unroll
            for (int kk = 0; kk < 4; ++kk) {
                v16h bfrag = load_b_f16(kb, D_, ct * 16, kk * 32, lane);
                acc = __builtin_amdgcn_wmma_f32_16x16x32_f16(
                    false, afrag[kk], false, bfrag, (short)0, acc, false, false);
            }
            int n = lane & 15, g = lane >> 4;
#pragma unroll
            for (int r = 0; r < 8; ++r)
                Sbuf[(rt * 16 + r + g * 8) * L_ + ct * 16 + n] = acc[r];
        }
    }
    __syncthreads();

    // Phase 2a: local term = 3x3 fc-weighted fold of raw scores, per block pair.
    float W[9];
#pragma unroll
    for (int i = 0; i < 9; ++i) W[i] = fc_w[i];
    const float fcb = fc_b[0];
    for (int i = 0; i < 16; ++i) {            // 16 q-blocks in panel
        float s = (float)D_ * fcb;            // j = tid (256 k-blocks)
#pragma unroll
        for (int a = 0; a < 3; ++a)
#pragma unroll
            for (int c = 0; c < 3; ++c)
                s += W[a * 3 + c] * Sbuf[(i * 3 + a) * L_ + tid * 3 + c];
        Lbuf[i * NBLK + tid] = s;
    }
    __syncthreads();

    // Phase 2b: softmax per row (one wave32 per row, 6 rows per wave).
    float* attn_b = attn_out + ((size_t)b * L_ + rowbase) * L_;
    for (int rr = 0; rr < 6; ++rr) {
        int r = wid * 6 + rr;
        const float* Srow = Sbuf + r * L_;
        const float* Lrow = Lbuf + (r / 3) * NBLK;
        float xv[24];
        float mx = -3.4e38f;
#pragma unroll
        for (int i = 0; i < 24; ++i) {
            int c = lane + i * 32;
            float x = (Srow[c] + Lrow[c / 3]) * INV_T;
            xv[i] = x;
            mx = fmaxf(mx, x);
        }
#pragma unroll
        for (int off = 16; off > 0; off >>= 1)
            mx = fmaxf(mx, __shfl_xor(mx, off, 32));
        float sum = 0.f;
#pragma unroll
        for (int i = 0; i < 24; ++i) {
            xv[i] = __expf(xv[i] - mx);
            sum += xv[i];
        }
#pragma unroll
        for (int off = 16; off > 0; off >>= 1)
            sum += __shfl_xor(sum, off, 32);
        float inv = __builtin_amdgcn_rcpf(sum);
#pragma unroll
        for (int i = 0; i < 24; ++i)
            attn_b[(size_t)r * L_ + lane + i * 32] = xv[i] * inv;
    }
}

// ---- Kernel 2: O = P * V. attn re-read lands in 192MB L2. ----
__global__ void __launch_bounds__(256) attn_pv_kernel(
    const float* __restrict__ attn, const _Float16* __restrict__ vt,
    float* __restrict__ out)
{
    const int blk  = blockIdx.x;
    const int b    = blk / 48;
    const int row0 = (blk % 48) * 16;
    const int lane = threadIdx.x & 31;
    const int wid  = threadIdx.x >> 5;
    const int n0   = wid * 16;            // 8 waves cover 128 d-columns

    const float*    Pb  = attn + ((size_t)b * L_ + row0) * L_;
    const _Float16* Vtb = vt + (size_t)b * D_ * L_;

    const int m = lane & 15, g = lane >> 4;
    v8f acc = {};
    for (int kk = 0; kk < 24; ++kk) {
        int k0 = kk * 32;
        // A fragment: probability row, f32 -> f16 in regs.
        const float* pa = Pb + (size_t)m * L_ + k0 + g * 16;
        v4f a0 = *(const v4f*)pa;
        v4f a1 = *(const v4f*)(pa + 4);
        v4f a2 = *(const v4f*)(pa + 8);
        v4f a3 = *(const v4f*)(pa + 12);
        v16h afrag;
#pragma unroll
        for (int i = 0; i < 4; ++i) {
            afrag[i]      = (_Float16)a0[i];
            afrag[i + 4]  = (_Float16)a1[i];
            afrag[i + 8]  = (_Float16)a2[i];
            afrag[i + 12] = (_Float16)a3[i];
        }
        v16h bfrag = *(const v16h*)(Vtb + (size_t)(n0 + m) * L_ + k0 + g * 16);
        acc = __builtin_amdgcn_wmma_f32_16x16x32_f16(
            false, afrag, false, bfrag, (short)0, acc, false, false);
    }
    float* ob = out + ((size_t)b * L_ + row0) * D_;
#pragma unroll
    for (int r = 0; r < 8; ++r)
        ob[(r + g * 8) * D_ + n0 + m] = acc[r];
}

extern "C" void kernel_launch(void* const* d_in, const int* in_sizes, int n_in,
                              void* d_out, int out_size, void* d_ws, size_t ws_size,
                              hipStream_t stream) {
    const float* q    = (const float*)d_in[0];
    const float* k    = (const float*)d_in[1];
    const float* v    = (const float*)d_in[2];
    const float* fc_w = (const float*)d_in[3];
    const float* fc_b = (const float*)d_in[4];

    float* out  = (float*)d_out;                       // (B, L, D)
    float* attn = out + (size_t)B_ * L_ * D_;          // (B, L, L)

    _Float16* qh = (_Float16*)d_ws;
    _Float16* kh = qh + (size_t)B_ * L_ * D_;
    _Float16* vt = kh + (size_t)B_ * L_ * D_;

    prep_kernel<<<(B_ * L_ * D_) / 256, 256, 0, stream>>>(q, k, v, qh, kh, vt);

    size_t smem = (size_t)(ROWS_PER_WG * L_ + 16 * NBLK) * sizeof(float); // 160 KB
    attn_scores_softmax_kernel<<<B_ * 16, 256, smem, stream>>>(qh, kh, fc_w, fc_b, attn);

    attn_pv_kernel<<<B_ * 48, 256, 0, stream>>>(attn, vt, out);
}
